// TopoGCN_50594714747412
// MI455X (gfx1250) — matrix operand order repaired and checked by reading, not compiled, not run on hardware
//
#include <hip/hip_runtime.h>
#include <hip/hip_bf16.h>
#include <math.h>

#define NN 100000
#define NE 1600000
#define NG 128
#define NFILT 8
#define HIDD 32

typedef __attribute__((ext_vector_type(16))) _Float16 v16h;
typedef __attribute__((ext_vector_type(8)))  float    v8f;

static const int TPB = 256;
static inline unsigned grid1(long long total) { return (unsigned)((total + TPB - 1) / TPB); }

// ---------- utility kernels ----------
__global__ void fill_f32_k(float* p, float v, long long n) {
    long long i = (long long)blockIdx.x * blockDim.x + threadIdx.x;
    if (i < n) p[i] = v;
}
__global__ void fill_u32_k(unsigned* p, unsigned v, long long n) {
    long long i = (long long)blockIdx.x * blockDim.x + threadIdx.x;
    if (i < n) p[i] = v;
}

// ---------- pack weights into WMMA B-fragment layout ----------
// Fragment layout: pack[((nt*KB + kb)*32 + lane)*16 + j]
//   col = nt*16 + (lane&15), k = kb*32 + (j&7) + (lane>>4)*8 + (j>>3)*16   (zero-padded cols)
__global__ void pack_b_k(const float* __restrict__ W, _Float16* __restrict__ pack,
                         int ncols, int ntiles, int KB) {
    int total = ntiles * KB * 32 * 16;
    int i = blockIdx.x * blockDim.x + threadIdx.x;
    if (i >= total) return;
    int j = i & 15;
    int lane = (i >> 4) & 31;
    int frag = i >> 9;  // nt*KB + kb
    int kb = frag % KB;
    int nt = frag / KB;
    int col = nt * 16 + (lane & 15);
    int kl = kb * 32 + (j & 7) + (lane >> 4) * 8 + (j >> 3) * 16;
    float v = (col < ncols) ? W[kl * ncols + col] : 0.0f;
    pack[i] = (_Float16)v;
}

// ---------- degree ----------
__global__ void count_edges_k(const int* __restrict__ dst, int* __restrict__ cnt, int e) {
    int i = blockIdx.x * blockDim.x + threadIdx.x;
    if (i < e) atomicAdd(&cnt[dst[i]], 1);
}
__global__ void dinv_k(const int* __restrict__ cnt, float* __restrict__ dinv, int n) {
    int i = blockIdx.x * blockDim.x + threadIdx.x;
    if (i < n) dinv[i] = rsqrtf(1.0f + (float)cnt[i]);
}

// ---------- layer 1 xw (K=1, trivial outer product) ----------
__global__ void layer1_xw_k(const float* __restrict__ x, const float* __restrict__ W1,
                            float* __restrict__ xw, long long total) {
    long long i = (long long)blockIdx.x * blockDim.x + threadIdx.x;
    if (i < total) xw[i] = x[i >> 5] * W1[i & 31];
}

// ---------- WMMA GEMM: C[rows,ncols] = A[rows,K] * Bpack (pre-packed fragments) ----------
// wave32; one wave per 16-row tile; K = KB*32; REQUIRES rows % 16 == 0.
template <int KB>
__global__ void gemm_wmma_k(const float* __restrict__ A, const _Float16* __restrict__ Bpack,
                            float* __restrict__ C, int rows, int ncols, int ntiles) {
    const int K = KB * 32;
    int wave = blockIdx.x * (blockDim.x >> 5) + (threadIdx.x >> 5);
    int lane = threadIdx.x & 31;
    if (wave * 16 >= rows) return;
    int half_id = lane >> 4;
    int m = lane & 15;
    int grow = wave * 16 + m;
    const float* Arow = A + (size_t)grow * K;

    // prefetch a future row tile of A (gfx1250 global_prefetch_b8)
    if (grow + 256 < rows) __builtin_prefetch(&A[(size_t)(grow + 256) * K], 0, 1);

    // A fragments: per-lane K runs are contiguous -> 4x float4 loads per fragment.
    v16h a[KB];
#pragma unroll
    for (int kb = 0; kb < KB; ++kb) {
        const float4* p0 = reinterpret_cast<const float4*>(Arow + kb * 32 + half_id * 8);
        const float4* p1 = reinterpret_cast<const float4*>(Arow + kb * 32 + 16 + half_id * 8);
        float4 c0 = p0[0], c1 = p0[1];
        float4 c2 = p1[0], c3 = p1[1];
        v16h af;
        af[0] = (_Float16)c0.x;  af[1] = (_Float16)c0.y;
        af[2] = (_Float16)c0.z;  af[3] = (_Float16)c0.w;
        af[4] = (_Float16)c1.x;  af[5] = (_Float16)c1.y;
        af[6] = (_Float16)c1.z;  af[7] = (_Float16)c1.w;
        af[8] = (_Float16)c2.x;  af[9] = (_Float16)c2.y;
        af[10] = (_Float16)c2.z; af[11] = (_Float16)c2.w;
        af[12] = (_Float16)c3.x; af[13] = (_Float16)c3.y;
        af[14] = (_Float16)c3.z; af[15] = (_Float16)c3.w;
        a[kb] = af;
    }

    const v16h* Bp = reinterpret_cast<const v16h*>(Bpack);
    for (int nt = 0; nt < ntiles; ++nt) {
        v8f acc = {};
#pragma unroll
        for (int kb = 0; kb < KB; ++kb) {
            v16h bf = Bp[(nt * KB + kb) * 32 + lane];  // one 32B vector load, branch-free
            acc = __builtin_amdgcn_wmma_f32_16x16x32_f16(
                false, a[kb], false, bf, (short)0, acc, false, false);
        }
        int gcol = nt * 16 + m;
        if (gcol < ncols) {  // only the ncols=8 call has a partial tile
            float* Cc = C + (size_t)(wave * 16 + half_id * 8) * ncols + gcol;
#pragma unroll
            for (int r = 0; r < 8; ++r) Cc[(size_t)r * ncols] = acc[r];
        }
    }
}

// ---------- edge scatter-sum with symmetric normalization ----------
__global__ void scatter_sum_k(const float* __restrict__ xw, const int* __restrict__ src,
                              const int* __restrict__ dst, const float* __restrict__ dinv,
                              float* __restrict__ agg, long long total, int F) {
    long long i = (long long)blockIdx.x * blockDim.x + threadIdx.x;
    if (i >= total) return;
    long long e = i / F;
    int f = (int)(i - e * F);
    int s = src[e], d = dst[e];
    float c = dinv[s] * dinv[d];
    atomicAdd(&agg[(size_t)d * F + f], xw[(size_t)s * F + f] * c);
}

// out = agg + xw*dinv^2 + b  (optional relu)
__global__ void gcn_final_k(const float* __restrict__ agg, const float* __restrict__ xw,
                            const float* __restrict__ dinv, const float* __restrict__ b,
                            float* __restrict__ out, long long total, int F, int relu) {
    long long i = (long long)blockIdx.x * blockDim.x + threadIdx.x;
    if (i >= total) return;
    long long n = i / F;
    int f = (int)(i - n * F);
    float dv = dinv[n];
    float v = agg[i] + xw[i] * dv * dv + b[f];
    out[i] = relu ? fmaxf(v, 0.0f) : v;
}

__global__ void add_bias_k(float* __restrict__ c, const float* __restrict__ b,
                           long long total, int F) {
    long long i = (long long)blockIdx.x * blockDim.x + threadIdx.x;
    if (i < total) c[i] += b[i % F];
}

// ---------- scatter-max (order-preserving uint encoding) ----------
__device__ __forceinline__ unsigned enc_f(float x) {
    unsigned u = __float_as_uint(x);
    return (u & 0x80000000u) ? ~u : (u | 0x80000000u);
}
__device__ __forceinline__ float dec_f(unsigned u) {
    return (u & 0x80000000u) ? __uint_as_float(u & 0x7FFFFFFFu) : __uint_as_float(~u);
}
#define ENC_NEG_INF 0x007FFFFFu  // enc(-inf)

__global__ void scatter_max_k(const float* __restrict__ f, const int* __restrict__ src,
                              const int* __restrict__ dst, unsigned* __restrict__ denc,
                              long long total) {
    long long i = (long long)blockIdx.x * blockDim.x + threadIdx.x;
    if (i >= total) return;
    long long e = i >> 3;
    int t = (int)(i & 7);
    int s = src[e], d = dst[e];
    atomicMax(&denc[(size_t)d * NFILT + t], enc_f(f[(size_t)s * NFILT + t]));
}
__global__ void resolve_death_k(const unsigned* __restrict__ denc, const float* __restrict__ f,
                                const int* __restrict__ cnt, float* __restrict__ death,
                                long long total) {
    long long i = (long long)blockIdx.x * blockDim.x + threadIdx.x;
    if (i >= total) return;
    long long n = i >> 3;
    death[i] = (cnt[n] > 0) ? dec_f(denc[i]) : f[i];
}

// tokens[n,t,c] = relu(f*Wp[0,c] + death*Wp[1,c] + bp[c])
__global__ void build_tokens_k(const float* __restrict__ f, const float* __restrict__ death,
                               const float* __restrict__ Wp, const float* __restrict__ bp,
                               float* __restrict__ tok, long long total) {
    long long i = (long long)blockIdx.x * blockDim.x + threadIdx.x;
    if (i >= total) return;
    long long nt = i >> 5;  // node*8 + filt
    int c = (int)(i & 31);
    float v = f[nt] * Wp[c] + death[nt] * Wp[32 + c] + bp[c];
    tok[i] = fmaxf(v, 0.0f);
}

// per-(node,filtration) attention over 8 tokens, HID=32; mean accumulated atomically
__global__ void attn_k(const float* __restrict__ q, const float* __restrict__ k,
                       const float* __restrict__ v, float* __restrict__ zmean, long long total) {
    long long i = (long long)blockIdx.x * blockDim.x + threadIdx.x;
    if (i >= total) return;
    long long n = i >> 3;
    const float* qf = q + (size_t)i * HIDD;
    float qr[HIDD];
#pragma unroll
    for (int c = 0; c < HIDD; ++c) qr[c] = qf[c];
    const float scale = 1.0f / sqrtf((float)HIDD);
    float sc[NFILT];
    float mx = -3.0e38f;
#pragma unroll
    for (int g = 0; g < NFILT; ++g) {
        const float* kg = k + ((size_t)n * NFILT + g) * HIDD;
        float s = 0.0f;
#pragma unroll
        for (int c = 0; c < HIDD; ++c) s += qr[c] * kg[c];
        s *= scale;
        sc[g] = s;
        mx = fmaxf(mx, s);
    }
    float den = 0.0f;
#pragma unroll
    for (int g = 0; g < NFILT; ++g) { sc[g] = expf(sc[g] - mx); den += sc[g]; }
    float z[HIDD];
#pragma unroll
    for (int c = 0; c < HIDD; ++c) z[c] = 0.0f;
#pragma unroll
    for (int g = 0; g < NFILT; ++g) {
        float w = sc[g] / den;
        const float* vg = v + ((size_t)n * NFILT + g) * HIDD;
#pragma unroll
        for (int c = 0; c < HIDD; ++c) z[c] += w * vg[c];
    }
#pragma unroll
    for (int c = 0; c < HIDD; ++c)
        atomicAdd(&zmean[(size_t)n * HIDD + c], z[c] * 0.125f);
}

// x2 = h1 + zWo + bo
__global__ void residual_k(const float* __restrict__ h1, const float* __restrict__ zWo,
                           const float* __restrict__ bo, float* __restrict__ x2, long long total) {
    long long i = (long long)blockIdx.x * blockDim.x + threadIdx.x;
    if (i < total) x2[i] = h1[i] + zWo[i] + bo[i & 31];
}

// ---------- pooling + FC ----------
__global__ void pool_sum_k(const float* __restrict__ h, const int* __restrict__ batch,
                           float* __restrict__ sums, long long total) {
    long long i = (long long)blockIdx.x * blockDim.x + threadIdx.x;
    if (i >= total) return;
    long long n = i >> 8;  // F=256
    int f = (int)(i & 255);
    atomicAdd(&sums[(size_t)batch[n] * 256 + f], h[i]);
}
__global__ void pool_cnt_k(const int* __restrict__ batch, int* __restrict__ gcnt, int n) {
    int i = blockIdx.x * blockDim.x + threadIdx.x;
    if (i < n) atomicAdd(&gcnt[batch[i]], 1);
}
__global__ void fc1_k(const float* __restrict__ sums, const int* __restrict__ gcnt,
                      const float* __restrict__ W, const float* __restrict__ b,
                      float* __restrict__ out) {
    int i = blockIdx.x * blockDim.x + threadIdx.x;  // 128*128
    if (i >= NG * 128) return;
    int gi = i >> 7, j = i & 127;
    float cm = fmaxf((float)gcnt[gi], 1.0f);
    float acc = b[j];
    for (int kk = 0; kk < 256; ++kk)
        acc += (sums[gi * 256 + kk] / cm) * W[kk * 128 + j];
    out[i] = fmaxf(acc, 0.0f);
}
__global__ void fc2_k(const float* __restrict__ g, const float* __restrict__ W,
                      const float* __restrict__ b, float* __restrict__ out) {
    int i = blockIdx.x * blockDim.x + threadIdx.x;  // 128*10
    if (i >= NG * 10) return;
    int gi = i / 10, j = i - gi * 10;
    float acc = b[j];
    for (int kk = 0; kk < 128; ++kk) acc += g[gi * 128 + kk] * W[kk * 10 + j];
    out[i] = acc;
}

// ---------- launch ----------
extern "C" void kernel_launch(void* const* d_in, const int* in_sizes, int n_in,
                              void* d_out, int out_size, void* d_ws, size_t ws_size,
                              hipStream_t stream) {
    (void)in_sizes; (void)n_in; (void)out_size; (void)ws_size;
    const float* x    = (const float*)d_in[0];
    const int*   ei   = (const int*)d_in[1];
    const int*   src  = ei;
    const int*   dst  = ei + NE;
    const int*   batch= (const int*)d_in[2];
    const float* W1 = (const float*)d_in[3];  const float* b1 = (const float*)d_in[4];
    const float* Wf = (const float*)d_in[5];  const float* bf = (const float*)d_in[6];
    const float* Wp = (const float*)d_in[7];  const float* bp = (const float*)d_in[8];
    const float* Wq = (const float*)d_in[9];  const float* Wk = (const float*)d_in[10];
    const float* Wv = (const float*)d_in[11];
    const float* Wo = (const float*)d_in[12]; const float* bo = (const float*)d_in[13];
    const float* W3 = (const float*)d_in[14]; const float* b3 = (const float*)d_in[15];
    const float* W4 = (const float*)d_in[16]; const float* b4 = (const float*)d_in[17];
    const float* Wfc1=(const float*)d_in[18]; const float* bfc1=(const float*)d_in[19];
    const float* Wfc2=(const float*)d_in[20]; const float* bfc2=(const float*)d_in[21];

    // workspace layout
    char* ws = (char*)d_ws;
    size_t off = 0;
    auto alloc = [&](size_t bytes) -> char* {
        char* p = ws + off;
        off += (bytes + 255) & ~(size_t)255;
        return p;
    };
    const size_t BIG = (size_t)NN * 256 * sizeof(float);  // 102.4 MB, == N*8*32 floats
    int*      cnt   = (int*)alloc((size_t)NN * 4);
    float*    dinv  = (float*)alloc((size_t)NN * 4);
    float*    h1    = (float*)alloc((size_t)NN * 32 * 4);
    float*    fbuf  = (float*)alloc((size_t)NN * NFILT * 4);
    unsigned* denc  = (unsigned*)alloc((size_t)NN * NFILT * 4);
    float*    death = (float*)alloc((size_t)NN * NFILT * 4);
    float*    zmean = (float*)alloc((size_t)NN * 32 * 4);
    float*    zWo   = (float*)alloc((size_t)NN * 32 * 4);
    float*    x2    = (float*)alloc((size_t)NN * 32 * 4);
    float*    B0    = (float*)alloc(BIG);
    float*    B1    = (float*)alloc(BIG);
    float*    B2    = (float*)alloc(BIG);
    float*    B3    = (float*)alloc(BIG);
    // packed B fragments (halfs): ntiles*KB*512
    _Float16* W3p   = (_Float16*)alloc((size_t)8 * 1 * 512 * 2);
    _Float16* W4p   = (_Float16*)alloc((size_t)16 * 4 * 512 * 2);
    _Float16* Wfp   = (_Float16*)alloc((size_t)1 * 1 * 512 * 2);
    _Float16* Wqp   = (_Float16*)alloc((size_t)2 * 1 * 512 * 2);
    _Float16* Wkp   = (_Float16*)alloc((size_t)2 * 1 * 512 * 2);
    _Float16* Wvp   = (_Float16*)alloc((size_t)2 * 1 * 512 * 2);
    _Float16* Wop   = (_Float16*)alloc((size_t)2 * 1 * 512 * 2);
    float*    sums  = (float*)alloc((size_t)NG * 256 * 4);
    int*      gcnt  = (int*)alloc((size_t)NG * 4);
    float*    gbuf  = (float*)alloc((size_t)NG * 128 * 4);

    // pack weights into WMMA B-fragment layout (small)
    pack_b_k<<<grid1(8 * 512), TPB, 0, stream>>>(W3, W3p, 128, 8, 1);
    pack_b_k<<<grid1(16 * 4 * 512), TPB, 0, stream>>>(W4, W4p, 256, 16, 4);
    pack_b_k<<<grid1(512), TPB, 0, stream>>>(Wf, Wfp, NFILT, 1, 1);
    pack_b_k<<<grid1(2 * 512), TPB, 0, stream>>>(Wq, Wqp, 32, 2, 1);
    pack_b_k<<<grid1(2 * 512), TPB, 0, stream>>>(Wk, Wkp, 32, 2, 1);
    pack_b_k<<<grid1(2 * 512), TPB, 0, stream>>>(Wv, Wvp, 32, 2, 1);
    pack_b_k<<<grid1(2 * 512), TPB, 0, stream>>>(Wo, Wop, 32, 2, 1);

    // degrees
    fill_u32_k<<<grid1(NN), TPB, 0, stream>>>((unsigned*)cnt, 0u, NN);
    count_edges_k<<<grid1(NE), TPB, 0, stream>>>(dst, cnt, NE);
    dinv_k<<<grid1(NN), TPB, 0, stream>>>(cnt, dinv, NN);

    const int WAVES_PER_BLOCK = 8;
    auto gemm_grid = [&](int rows) {
        int tiles = (rows + 15) >> 4;
        return (unsigned)((tiles + WAVES_PER_BLOCK - 1) / WAVES_PER_BLOCK);
    };

    // ---- GCN layer 1 (K=1 outer product, F=32) ----
    layer1_xw_k<<<grid1((long long)NN * 32), TPB, 0, stream>>>(x, W1, B0, (long long)NN * 32);
    fill_f32_k<<<grid1((long long)NN * 32), TPB, 0, stream>>>(B1, 0.0f, (long long)NN * 32);
    scatter_sum_k<<<grid1((long long)NE * 32), TPB, 0, stream>>>(B0, src, dst, dinv, B1,
                                                                 (long long)NE * 32, 32);
    gcn_final_k<<<grid1((long long)NN * 32), TPB, 0, stream>>>(B1, B0, dinv, b1, h1,
                                                               (long long)NN * 32, 32, 1);

    // ---- TOGL: filtrations f = h1 @ Wf + bf ----
    gemm_wmma_k<1><<<gemm_grid(NN), TPB, 0, stream>>>(h1, Wfp, fbuf, NN, NFILT, 1);
    add_bias_k<<<grid1((long long)NN * NFILT), TPB, 0, stream>>>(fbuf, bf,
                                                                 (long long)NN * NFILT, NFILT);
    fill_u32_k<<<grid1((long long)NN * NFILT), TPB, 0, stream>>>(denc, ENC_NEG_INF,
                                                                 (long long)NN * NFILT);
    scatter_max_k<<<grid1((long long)NE * NFILT), TPB, 0, stream>>>(fbuf, src, dst, denc,
                                                                    (long long)NE * NFILT);
    resolve_death_k<<<grid1((long long)NN * NFILT), TPB, 0, stream>>>(denc, fbuf, cnt, death,
                                                                      (long long)NN * NFILT);
    build_tokens_k<<<grid1((long long)NN * NFILT * 32), TPB, 0, stream>>>(
        fbuf, death, Wp, bp, B0, (long long)NN * NFILT * 32);

    // q,k,v over flattened tokens [N*8, 32]
    gemm_wmma_k<1><<<gemm_grid(NN * NFILT), TPB, 0, stream>>>(B0, Wqp, B1, NN * NFILT, 32, 2);
    gemm_wmma_k<1><<<gemm_grid(NN * NFILT), TPB, 0, stream>>>(B0, Wkp, B2, NN * NFILT, 32, 2);
    gemm_wmma_k<1><<<gemm_grid(NN * NFILT), TPB, 0, stream>>>(B0, Wvp, B3, NN * NFILT, 32, 2);

    fill_f32_k<<<grid1((long long)NN * 32), TPB, 0, stream>>>(zmean, 0.0f, (long long)NN * 32);
    attn_k<<<grid1((long long)NN * NFILT), TPB, 0, stream>>>(B1, B2, B3, zmean,
                                                             (long long)NN * NFILT);
    gemm_wmma_k<1><<<gemm_grid(NN), TPB, 0, stream>>>(zmean, Wop, zWo, NN, 32, 2);
    residual_k<<<grid1((long long)NN * 32), TPB, 0, stream>>>(h1, zWo, bo, x2,
                                                              (long long)NN * 32);

    // ---- GCN layer 2: x2 @ W3 [32->128] ----
    gemm_wmma_k<1><<<gemm_grid(NN), TPB, 0, stream>>>(x2, W3p, B0, NN, 128, 8);
    fill_f32_k<<<grid1((long long)NN * 128), TPB, 0, stream>>>(B1, 0.0f, (long long)NN * 128);
    scatter_sum_k<<<grid1((long long)NE * 128), TPB, 0, stream>>>(B0, src, dst, dinv, B1,
                                                                  (long long)NE * 128, 128);
    gcn_final_k<<<grid1((long long)NN * 128), TPB, 0, stream>>>(B1, B0, dinv, b3, B2,
                                                                (long long)NN * 128, 128, 1);

    // ---- GCN layer 3: h2 @ W4 [128->256] ----
    gemm_wmma_k<4><<<gemm_grid(NN), TPB, 0, stream>>>(B2, W4p, B0, NN, 256, 16);
    fill_f32_k<<<grid1((long long)NN * 256), TPB, 0, stream>>>(B1, 0.0f, (long long)NN * 256);
    scatter_sum_k<<<grid1((long long)NE * 256), TPB, 0, stream>>>(B0, src, dst, dinv, B1,
                                                                  (long long)NE * 256, 256);
    gcn_final_k<<<grid1((long long)NN * 256), TPB, 0, stream>>>(B1, B0, dinv, b4, B3,
                                                                (long long)NN * 256, 256, 1);

    // ---- global mean pool + MLP head ----
    fill_f32_k<<<grid1((long long)NG * 256), TPB, 0, stream>>>(sums, 0.0f, (long long)NG * 256);
    fill_u32_k<<<grid1(NG), TPB, 0, stream>>>((unsigned*)gcnt, 0u, NG);
    pool_sum_k<<<grid1((long long)NN * 256), TPB, 0, stream>>>(B3, batch, sums,
                                                               (long long)NN * 256);
    pool_cnt_k<<<grid1(NN), TPB, 0, stream>>>(batch, gcnt, NN);
    fc1_k<<<grid1(NG * 128), TPB, 0, stream>>>(sums, gcnt, Wfc1, bfc1, gbuf);
    fc2_k<<<grid1(NG * 10), TPB, 0, stream>>>(gbuf, Wfc2, bfc2, (float*)d_out);
}
